// GrapsuleNet_60601988546906
// MI455X (gfx1250) — compile-verified, matrix operand order checked by reading, not run on hardware
//
#include <hip/hip_runtime.h>
#include <cstdint>
#include <cstddef>

// ---------------- problem constants ----------------
#define BB   32
#define CC   384
#define NN   4096      // 64*64
#define EE   256
#define NH   16
#define HD   16
#define HID  128
#define RFFH 32
#define NEDGE 7680     // B * NH*(NH-1)
#define NNODE 512      // B * NH
#define NKC  12        // 384/32 k-steps
#define NCHUNK 128     // pixels per workgroup chunk
#define NCHUNKS 32     // 4096/128

typedef __attribute__((ext_vector_type(16))) __bf16 v16bf;
typedef __attribute__((ext_vector_type(8)))  float  v8f;

union AFrag { unsigned u[8]; v16bf v; };

__device__ __forceinline__ unsigned f2bf2(float lo, float hi) {
    unsigned a = __float_as_uint(lo);
    unsigned b = __float_as_uint(hi);
    a = (a + 0x7FFFu + ((a >> 16) & 1u)) >> 16;
    b = (b + 0x7FFFu + ((b >> 16) & 1u)) >> 16;
    return (a & 0xFFFFu) | (b << 16);
}
__device__ __forceinline__ unsigned short f2bf(float f) {
    unsigned a = __float_as_uint(f);
    a = (a + 0x7FFFu + ((a >> 16) & 1u)) >> 16;
    return (unsigned short)a;
}
__device__ __forceinline__ v8f wmma_bf16(const AFrag& A, const AFrag& B, v8f C) {
    return __builtin_amdgcn_wmma_f32_16x16x32_bf16(false, A.v, false, B.v,
                                                   (short)0, C, false, false);
}

// ---------------- K0: pack kw/vw into bf16 A-fragment layout ----------------
// word[((s*12+kc)*32+L)*8 + j] = bf16pair(kw[e][c], kw[e][c+1])
//   e = s*16 + (L&15), c = kc*32 + ((j>=4)?16:0) + ((L>=16)?8:0) + 2*(j&3)
__global__ __launch_bounds__(256) void k_pack(const float* __restrict__ kw,
                                              const float* __restrict__ vw,
                                              unsigned* __restrict__ kwp,
                                              unsigned* __restrict__ vwp) {
    int id = blockIdx.x * 256 + threadIdx.x;   // 2*49152
    const float* w; unsigned* o; int t = id;
    if (t < 49152) { w = kw; o = kwp; } else { w = vw; o = vwp; t -= 49152; }
    int j  = t & 7;
    int L  = (t >> 3) & 31;
    int kc = (t >> 8) % NKC;
    int s  = (t >> 8) / NKC;
    int e  = s * 16 + (L & 15);
    int c  = kc * 32 + ((j >= 4) ? 16 : 0) + ((L >= 16) ? 8 : 0) + 2 * (j & 3);
    o[t] = f2bf2(w[e * CC + c], w[e * CC + c + 1]);
}

// ---------------- K1: per-(b,c) mean over pixels ----------------
__global__ __launch_bounds__(256) void k_mean(const float* __restrict__ x,
                                              float* __restrict__ xm) {
    __shared__ float red[256];
    int row = blockIdx.x;                      // b*CC + c
    const float* p = x + (size_t)row * NN;
    float s = 0.f;
    for (int i = threadIdx.x; i < NN; i += 256) s += p[i];
    red[threadIdx.x] = s; __syncthreads();
    for (int st = 128; st > 0; st >>= 1) {
        if (threadIdx.x < st) red[threadIdx.x] += red[threadIdx.x + st];
        __syncthreads();
    }
    if (threadIdx.x == 0) xm[row] = red[0] * (1.0f / (float)NN);
}

// ---------------- K2: q = xm @ qw.T + qb ----------------
__global__ __launch_bounds__(256) void k_q(const float* __restrict__ xm,
                                           const float* __restrict__ qw,
                                           const float* __restrict__ qb,
                                           float* __restrict__ q) {
    int id = blockIdx.x * 256 + threadIdx.x;   // 8192
    int b = id >> 8, e = id & 255;
    const float* xr = xm + b * CC;
    const float* wr = qw + e * CC;
    float s = 0.f;
    for (int c = 0; c < CC; ++c) s += xr[c] * wr[c];
    q[id] = s + qb[e];
}

// ---------------- K3: fused k-GEMM + q.k -> logits ----------------
// grid (B, NCHUNKS), 256 threads (8 waves). Wave w owns heads 2w, 2w+1.
// Software-pipelined: x(kc+1) + A(kc+1) loads overlap WMMAs of kc.
__global__ __launch_bounds__(256, 1) void k_logits(const float* __restrict__ x,
                                                   const unsigned* __restrict__ kwp,
                                                   const float* __restrict__ q,
                                                   float* __restrict__ logits) {
    __shared__ unsigned xs[2 * 2048];          // double-buffered 32c x 128n bf16 pairs
    int b = blockIdx.x, chunk = blockIdx.y;
    int n0 = chunk * NCHUNK;
    int tid = threadIdx.x, lane = tid & 31, wave = tid >> 5;
    const float* xb = x + (size_t)b * CC * NN;

    float2 xreg[4][2];
    auto loadX = [&](int kc) {
#pragma unroll
        for (int it = 0; it < 4; ++it) {
            int i = tid + it * 256;            // 0..1023
            int n2 = (i & 63) * 2, cp = i >> 6;
            const float* base = xb + (size_t)(kc * 32 + cp * 2) * NN + n0 + n2;
            xreg[it][0] = *(const float2*)base;
            xreg[it][1] = *(const float2*)(base + NN);
        }
    };
    auto storeX = [&](int buf) {
#pragma unroll
        for (int it = 0; it < 4; ++it) {
            int i = tid + it * 256;
            int n2 = (i & 63) * 2, cp = i >> 6;
            uint2 w;
            w.x = f2bf2(xreg[it][0].x, xreg[it][1].x);
            w.y = f2bf2(xreg[it][0].y, xreg[it][1].y);
            *(uint2*)&xs[buf * 2048 + cp * 128 + n2] = w;
        }
    };
    auto loadA = [&](int kc, AFrag& f0, AFrag& f1) {
        const unsigned* p0 = kwp + (((size_t)(2 * wave + 0) * NKC + kc) * 32 + lane) * 8;
        const unsigned* p1 = kwp + (((size_t)(2 * wave + 1) * NKC + kc) * 32 + lane) * 8;
#pragma unroll
        for (int j = 0; j < 8; ++j) { f0.u[j] = p0[j]; f1.u[j] = p1[j]; }
    };

    v8f acc[2][8];
    v8f z = {0.f, 0.f, 0.f, 0.f, 0.f, 0.f, 0.f, 0.f};
#pragma unroll
    for (int s = 0; s < 2; ++s)
#pragma unroll
        for (int nt = 0; nt < 8; ++nt) acc[s][nt] = z;

    AFrag a0, a1, na0, na1;
    loadX(0);
    storeX(0);
    loadA(0, a0, a1);
    int kadd = (lane >= 16) ? 4 : 0;

    for (int kc = 0; kc < NKC; ++kc) {
        int buf = kc & 1;
        __syncthreads();                       // xs[buf] ready
        if (kc + 1 < NKC) { loadX(kc + 1); loadA(kc + 1, na0, na1); }
#pragma unroll
        for (int nt = 0; nt < 8; ++nt) {
            AFrag bf;
            int ncol = nt * 16 + (lane & 15);
#pragma unroll
            for (int j = 0; j < 8; ++j) {
                int kp = ((j >= 4) ? 8 : 0) + kadd + (j & 3);
                bf.u[j] = xs[buf * 2048 + kp * 128 + ncol];
            }
            acc[0][nt] = wmma_bf16(a0, bf, acc[0][nt]);
            acc[1][nt] = wmma_bf16(a1, bf, acc[1][nt]);
        }
        if (kc + 1 < NKC) { storeX(buf ^ 1); a0 = na0; a1 = na1; }
    }
    // epilogue: logits[b][n][h] = (1/4) * sum_d q[b,h,d] * K[h*16+d][n]
#pragma unroll
    for (int s = 0; s < 2; ++s) {
        int h = 2 * wave + s;
        float qv[8];
        const float* qr = q + b * EE + h * HD + ((lane >= 16) ? 8 : 0);
#pragma unroll
        for (int r = 0; r < 8; ++r) qv[r] = qr[r];
#pragma unroll
        for (int nt = 0; nt < 8; ++nt) {
            float sum = 0.f;
#pragma unroll
            for (int r = 0; r < 8; ++r) sum += qv[r] * acc[s][nt][r];
            sum *= 0.25f;
            sum += __shfl_xor(sum, 16, 32);
            if (lane < 16) {
                int n = n0 + nt * 16 + lane;
                logits[((size_t)b * NN + n) * NH + h] = sum;
            }
        }
    }
}

// ---------------- K4: softmax over pixels + attention-weighted pos ----------
__global__ __launch_bounds__(256) void k_softmax(const float* __restrict__ logits,
                                                 float* __restrict__ attn,
                                                 float* __restrict__ pos) {
    __shared__ float red[256];
    int d = blockIdx.x, b = d >> 4, h = d & 15;
    const float* lp = logits + (size_t)b * NN * NH + h;
    float* ap = attn + (size_t)b * NN * NH + h;
    int tid = threadIdx.x;

    float mx = -1e30f;
    for (int n = tid; n < NN; n += 256) mx = fmaxf(mx, lp[(size_t)n * NH]);
    red[tid] = mx; __syncthreads();
    for (int st = 128; st > 0; st >>= 1) { if (tid < st) red[tid] = fmaxf(red[tid], red[tid + st]); __syncthreads(); }
    mx = red[0]; __syncthreads();

    float sm = 0.f;
    for (int n = tid; n < NN; n += 256) sm += __expf(lp[(size_t)n * NH] - mx);
    red[tid] = sm; __syncthreads();
    for (int st = 128; st > 0; st >>= 1) { if (tid < st) red[tid] += red[tid + st]; __syncthreads(); }
    float inv = 1.0f / red[0]; __syncthreads();

    float px = 0.f, py = 0.f;
    for (int n = tid; n < NN; n += 256) {
        float a = __expf(lp[(size_t)n * NH] - mx) * inv;
        ap[(size_t)n * NH] = a;
        px += a * (float)(n >> 6);
        py += a * (float)(n & 63);
    }
    red[tid] = px; __syncthreads();
    for (int st = 128; st > 0; st >>= 1) { if (tid < st) red[tid] += red[tid + st]; __syncthreads(); }
    if (tid == 0) pos[d * 2 + 0] = red[0];
    __syncthreads();
    red[tid] = py; __syncthreads();
    for (int st = 128; st > 0; st >>= 1) { if (tid < st) red[tid] += red[tid + st]; __syncthreads(); }
    if (tid == 0) pos[d * 2 + 1] = red[0];
}

// ---------------- K5: fused v-GEMM + attnT@V -> per-chunk partials ----------
// dynamic LDS: vs[16384] + st[2*2048] words (x double-buffer / attn staging)
__global__ __launch_bounds__(256, 1) void k_values_part(const float* __restrict__ x,
                                                        const unsigned* __restrict__ vwp,
                                                        const float* __restrict__ attn,
                                                        float* __restrict__ pOut) {
    extern __shared__ unsigned smem[];
    unsigned* vs = smem;                    // 64 n-pairs x 256 e  (bf16 pairs)
    unsigned* st = smem + 16384;            // staging (double buffer / attn f32)
    float* stF = (float*)st;
    unsigned short* vsh = (unsigned short*)vs;

    int b = blockIdx.x, chunk = blockIdx.y;
    int n0 = chunk * NCHUNK;
    int tid = threadIdx.x, lane = tid & 31, wave = tid >> 5;
    const float* xb = x + (size_t)b * CC * NN;

    float2 xreg[4][2];
    auto loadX = [&](int kc) {
#pragma unroll
        for (int it = 0; it < 4; ++it) {
            int i = tid + it * 256;
            int n2 = (i & 63) * 2, cp = i >> 6;
            const float* base = xb + (size_t)(kc * 32 + cp * 2) * NN + n0 + n2;
            xreg[it][0] = *(const float2*)base;
            xreg[it][1] = *(const float2*)(base + NN);
        }
    };
    auto storeX = [&](int buf) {
#pragma unroll
        for (int it = 0; it < 4; ++it) {
            int i = tid + it * 256;
            int n2 = (i & 63) * 2, cp = i >> 6;
            uint2 w;
            w.x = f2bf2(xreg[it][0].x, xreg[it][1].x);
            w.y = f2bf2(xreg[it][0].y, xreg[it][1].y);
            *(uint2*)&st[buf * 2048 + cp * 128 + n2] = w;
        }
    };
    auto loadA = [&](int kc, AFrag& f0, AFrag& f1) {
        const unsigned* p0 = vwp + (((size_t)(2 * wave + 0) * NKC + kc) * 32 + lane) * 8;
        const unsigned* p1 = vwp + (((size_t)(2 * wave + 1) * NKC + kc) * 32 + lane) * 8;
#pragma unroll
        for (int j = 0; j < 8; ++j) { f0.u[j] = p0[j]; f1.u[j] = p1[j]; }
    };

    v8f z = {0.f, 0.f, 0.f, 0.f, 0.f, 0.f, 0.f, 0.f};
    v8f acc[2][8];
#pragma unroll
    for (int s = 0; s < 2; ++s)
#pragma unroll
        for (int nt = 0; nt < 8; ++nt) acc[s][nt] = z;

    // ---- phase A: V[e][n] chunk via WMMA (software-pipelined) ----
    AFrag a0, a1, na0, na1;
    loadX(0);
    storeX(0);
    loadA(0, a0, a1);
    int kadd = (lane >= 16) ? 4 : 0;

    for (int kc = 0; kc < NKC; ++kc) {
        int buf = kc & 1;
        __syncthreads();
        if (kc + 1 < NKC) { loadX(kc + 1); loadA(kc + 1, na0, na1); }
#pragma unroll
        for (int nt = 0; nt < 8; ++nt) {
            AFrag bf;
            int ncol = nt * 16 + (lane & 15);
#pragma unroll
            for (int j = 0; j < 8; ++j) {
                int kp = ((j >= 4) ? 8 : 0) + kadd + (j & 3);
                bf.u[j] = st[buf * 2048 + kp * 128 + ncol];
            }
            acc[0][nt] = wmma_bf16(a0, bf, acc[0][nt]);
            acc[1][nt] = wmma_bf16(a1, bf, acc[1][nt]);
        }
        if (kc + 1 < NKC) { storeX(buf ^ 1); a0 = na0; a1 = na1; }
    }
    // store V to LDS as bf16, pairs along n for phase-B B-fragments
#pragma unroll
    for (int s = 0; s < 2; ++s) {
        int es0 = (2 * wave + s) * 16 + ((lane >= 16) ? 8 : 0);
#pragma unroll
        for (int nt = 0; nt < 8; ++nt) {
            int n = nt * 16 + (lane & 15);
            int np = n >> 1, hs = n & 1;
#pragma unroll
            for (int r = 0; r < 8; ++r) {
                int e = es0 + r;
                vsh[(size_t)(np * 256 + e) * 2 + hs] = f2bf(acc[s][nt][r]);
            }
        }
    }
    __syncthreads();

    // ---- stage attn chunk [128n x 16h] via CDNA5 async global->LDS copy ----
    for (int i = tid; i < 2048; i += 256) {
        int n = i >> 4, h = i & 15;
        const float* gp = attn + ((size_t)b * NN + n0 + n) * NH + h;
        unsigned ldsoff = (unsigned)(uintptr_t)(stF + i);   // low 32 bits = LDS byte offset
        asm volatile("global_load_async_to_lds_b32 %0, %1, off"
                     :: "v"(ldsoff), "v"((unsigned long long)(uintptr_t)gp)
                     : "memory");
    }
    asm volatile("s_wait_asynccnt 0x0" ::: "memory");
    __syncthreads();

    // ---- phase B: out[16h x 16e] += attnT @ V, wave w owns e-tiles 2w,2w+1 ----
    v8f oc[2]; oc[0] = z; oc[1] = z;
    for (int ks = 0; ks < 4; ++ks) {
        AFrag af;
        int hrow = lane & 15;
#pragma unroll
        for (int j = 0; j < 8; ++j) {
            int kp = ((j >= 4) ? 8 : 0) + kadd + (j & 3);
            int n2 = ks * 32 + 2 * kp;
            af.u[j] = f2bf2(stF[n2 * 16 + hrow], stF[(n2 + 1) * 16 + hrow]);
        }
#pragma unroll
        for (int et = 0; et < 2; ++et) {
            AFrag bf;
            int ecol = (2 * wave + et) * 16 + (lane & 15);
#pragma unroll
            for (int j = 0; j < 8; ++j) {
                int kp = ((j >= 4) ? 8 : 0) + kadd + (j & 3);
                bf.u[j] = vs[(ks * 16 + kp) * 256 + ecol];
            }
            oc[et] = wmma_bf16(af, bf, oc[et]);
        }
    }
    // write partials pOut[b][chunk][h][e]
#pragma unroll
    for (int et = 0; et < 2; ++et) {
        int e = (2 * wave + et) * 16 + (lane & 15);
        int hb = (lane >= 16) ? 8 : 0;
#pragma unroll
        for (int r = 0; r < 8; ++r) {
            int h = r + hb;
            pOut[(((size_t)b * NCHUNKS + chunk) * NH + h) * EE + e] = oc[et][r];
        }
    }
}

// ---------------- K5b: reduce chunk partials + vb -> values ----------------
__global__ __launch_bounds__(256) void k_values_reduce(const float* __restrict__ pOut,
                                                       const float* __restrict__ vb,
                                                       float* __restrict__ values) {
    int id = blockIdx.x * 256 + threadIdx.x;   // 131072
    int e = id & 255, d = id >> 8, b = d >> 4, h = d & 15;
    float s = vb[e];
    for (int c = 0; c < NCHUNKS; ++c)
        s += pOut[(((size_t)b * NCHUNKS + c) * NH + h) * EE + e];
    values[id] = s;
}

// ---------------- K6a: P = values @ msg_w.T + msg_b (per node) -------------
__global__ __launch_bounds__(128) void k_P(const float* __restrict__ values,
                                           const float* __restrict__ msg_w,
                                           const float* __restrict__ msg_b,
                                           float* __restrict__ P) {
    __shared__ float vrow[EE];
    int i = blockIdx.x, t = threadIdx.x;
    for (int e = t; e < EE; e += 128) vrow[e] = values[(size_t)i * EE + e];
    __syncthreads();
    float s = msg_b[t];
    const float* w = msg_w + (size_t)t * EE;
    for (int e = 0; e < EE; ++e) s += vrow[e] * w[e];
    P[(size_t)i * HID + t] = s;
}

// ---------------- K6b: per-edge RFF gate * P[src] -> m ---------------------
__global__ __launch_bounds__(128) void k_edges(const float* __restrict__ pos,
                                               const float* __restrict__ B_rff,
                                               const float* __restrict__ emb_w,
                                               const float* __restrict__ P,
                                               const int* __restrict__ esrc,
                                               const int* __restrict__ edst,
                                               float* __restrict__ m) {
    __shared__ float ea[2 * RFFH];
    int e = blockIdx.x, t = threadIdx.x;
    int s = esrc[e], d0 = edst[e];
    float rx = pos[s * 2 + 0] - pos[d0 * 2 + 0];
    float ry = pos[s * 2 + 1] - pos[d0 * 2 + 1];
    if (t < RFFH) {
        float pr = rx * B_rff[t * 2 + 0] + ry * B_rff[t * 2 + 1];
        ea[t]        = 1.41421356237309515f * sinf(pr);
        ea[t + RFFH] = 1.41421356237309515f * cosf(pr);
    }
    __syncthreads();
    float g = 0.f;
    const float* w = emb_w + (size_t)t * (2 * RFFH);
    for (int a = 0; a < 2 * RFFH; ++a) g += ea[a] * w[a];
    m[(size_t)e * HID + t] = P[(size_t)s * HID + t] * g;
}

// ---------------- K7: deterministic mean aggregation per dst ---------------
__global__ __launch_bounds__(128) void k_agg(const float* __restrict__ m,
                                             float* __restrict__ agg) {
    int d = blockIdx.x, t = threadIdx.x;
    int b = d >> 4, j = d & 15;
    float s = 0.f;
    for (int i = 0; i < NH; ++i) {
        if (i == j) continue;
        int p = i * 15 + (j - ((j > i) ? 1 : 0));     // edge (i->j) index in batch
        s += m[(size_t)(b * 240 + p) * HID + t];
    }
    agg[(size_t)d * HID + t] = s * (1.0f / 15.0f);
}

// ---------------- K8: MLP + layer scale + skip -> out ----------------------
__global__ __launch_bounds__(256) void k_mlp(const float* __restrict__ agg,
                                             const float* __restrict__ lin1_w,
                                             const float* __restrict__ lin1_b,
                                             const float* __restrict__ lin2_w,
                                             const float* __restrict__ lin2_b,
                                             const float* __restrict__ layer_scale,
                                             const float* __restrict__ values,
                                             float* __restrict__ out) {
    __shared__ float ar[HID];
    __shared__ float hb[4 * HID];
    int d = blockIdx.x, t = threadIdx.x;
    if (t < HID) ar[t] = agg[(size_t)d * HID + t];
    __syncthreads();
    for (int u = t; u < 4 * HID; u += 256) {
        float s = lin1_b[u];
        const float* w = lin1_w + (size_t)u * HID;
        for (int c = 0; c < HID; ++c) s += ar[c] * w[c];
        hb[u] = s / (1.0f + __expf(-s));               // silu
    }
    __syncthreads();
    float s = lin2_b[t];
    const float* w = lin2_w + (size_t)t * (4 * HID);
    for (int u = 0; u < 4 * HID; ++u) s += hb[u] * w[u];
    out[(size_t)d * EE + t] = layer_scale[t] * s + values[(size_t)d * EE + t];
}

// ---------------- host ----------------
extern "C" void kernel_launch(void* const* d_in, const int* in_sizes, int n_in,
                              void* d_out, int out_size, void* d_ws, size_t ws_size,
                              hipStream_t stream) {
    const float* x      = (const float*)d_in[0];
    const float* qw     = (const float*)d_in[1];
    const float* qb     = (const float*)d_in[2];
    const float* kw     = (const float*)d_in[3];
    const float* kb     = (const float*)d_in[4];  (void)kb; // zero / softmax-invariant shift
    const float* vw     = (const float*)d_in[5];
    const float* vb     = (const float*)d_in[6];
    const float* B_rff  = (const float*)d_in[7];
    const float* msg_w  = (const float*)d_in[8];
    const float* msg_b  = (const float*)d_in[9];
    const float* emb_w  = (const float*)d_in[10];
    const float* lin1_w = (const float*)d_in[11];
    const float* lin1_b = (const float*)d_in[12];
    const float* lin2_w = (const float*)d_in[13];
    const float* lin2_b = (const float*)d_in[14];
    const float* lscale = (const float*)d_in[15];
    const int*   esrc   = (const int*)d_in[16];
    const int*   edst   = (const int*)d_in[17];
    float* out = (float*)d_out;

    // workspace carve-up (bytes), 256-aligned
    char* w = (char*)d_ws;
    size_t off = 0;
    auto take = [&](size_t bytes) { void* p = w + off; off = (off + bytes + 255) & ~(size_t)255; return p; };
    unsigned* kwp   = (unsigned*)take(49152 * 4);
    unsigned* vwp   = (unsigned*)take(49152 * 4);
    float* xm       = (float*)take((size_t)BB * CC * 4);
    float* q        = (float*)take((size_t)BB * EE * 4);
    float* logits   = (float*)take((size_t)BB * NN * NH * 4);  // softmax in-place -> attn
    float* pos      = (float*)take((size_t)NNODE * 2 * 4);
    float* pOut     = (float*)take((size_t)BB * NCHUNKS * NH * EE * 4);
    float* values   = (float*)take((size_t)NNODE * EE * 4);
    float* P        = (float*)take((size_t)NNODE * HID * 4);
    float* m        = (float*)take((size_t)NEDGE * HID * 4);
    float* agg      = (float*)take((size_t)NNODE * HID * 4);
    (void)ws_size; (void)in_sizes; (void)n_in; (void)out_size;

    k_pack<<<384, 256, 0, stream>>>(kw, vw, kwp, vwp);
    k_mean<<<BB * CC, 256, 0, stream>>>(x, xm);
    k_q<<<BB * EE / 256, 256, 0, stream>>>(xm, qw, qb, q);
    k_logits<<<dim3(BB, NCHUNKS), 256, 0, stream>>>(x, kwp, q, logits);
    k_softmax<<<NNODE, 256, 0, stream>>>(logits, /*attn in-place*/ logits, pos);
    k_values_part<<<dim3(BB, NCHUNKS), 256, (16384 + 2 * 2048) * 4, stream>>>(x, vwp, logits, pOut);
    k_values_reduce<<<NNODE, 256, 0, stream>>>(pOut, vb, values);
    k_P<<<NNODE, 128, 0, stream>>>(values, msg_w, msg_b, P);
    k_edges<<<NEDGE, 128, 0, stream>>>(pos, B_rff, emb_w, P, esrc, edst, m);
    k_agg<<<NNODE, 128, 0, stream>>>(m, agg);
    k_mlp<<<NNODE, 256, 0, stream>>>(agg, lin1_w, lin1_b, lin2_w, lin2_b, lscale, values, out);
}
